// BrickVectorEdgeModel_21406117003470
// MI455X (gfx1250) — compile-verified, matrix operand order checked by reading, not run on hardware
//
#include <hip/hip_runtime.h>

// ---------------------------------------------------------------------------
// BrickVectorEdgeModel on gfx1250: bf16 WMMA (f32 accumulate), fused edge MLP.
// Edge kernel: 4 waves/WG cooperatively double-buffer 16KB weight tiles in LDS
// (4x less L2 weight traffic); activations live in wave-private LDS ping-pong.
// ---------------------------------------------------------------------------

typedef __bf16        v16bf __attribute__((ext_vector_type(16)));
typedef float         v8f   __attribute__((ext_vector_type(8)));
typedef unsigned int  v4u   __attribute__((ext_vector_type(4)));

#define DIN    256
#define HD     512
#define NBRICK 128
#define NBATCH 8
#define MTOT   (NBRICK * NBATCH)   // 1024 node rows

// ----------------------------- helpers -------------------------------------

static __device__ __forceinline__ unsigned short f2bf(float f) {
  union { float f; unsigned int u; } c; c.f = f;
  unsigned int r = c.u + 0x7FFFu + ((c.u >> 16) & 1u);   // round-to-nearest-even
  return (unsigned short)(r >> 16);
}

// Load one 16x32 bf16 fragment for wmma_f32_16x16x32_bf16 (ISA 7.12.2 layout):
// lanes 0-15 hold K 0..7 / 16..23 of row (lane); lanes 16-31 hold K 8..15 /
// 24..31 of row (lane-16).  Caller pre-applies row/lane-half offsets; p must
// be 16B aligned.
static __device__ __forceinline__ v16bf load_frag(const unsigned short* p) {
  union { v16bf v; v4u u[2]; } f;
  f.u[0] = *(const v4u*)(p);
  f.u[1] = *(const v4u*)(p + 16);
  return f.v;
}

static __device__ __forceinline__ v8f wmma_bf16(v16bf a, v16bf b, v8f c) {
  return __builtin_amdgcn_wmma_f32_16x16x32_bf16(
      /*neg_a=*/false, a, /*neg_b=*/false, b,
      /*c_mod=*/(short)0, c, /*reuse_a=*/false, /*reuse_b=*/false);
}

// ----------------------------- prep kernels --------------------------------

__global__ void cast_k(const float* __restrict__ src,
                       unsigned short* __restrict__ dst, int total) {
  for (int i = blockIdx.x * blockDim.x + threadIdx.x; i < total;
       i += gridDim.x * blockDim.x)
    dst[i] = f2bf(src[i]);
}

// dst[n*K + k] = bf16(src[(row0 + k)*ld + n])   (transpose KxN -> NxK)
__global__ void cast_transpose_k(const float* __restrict__ src,
                                 unsigned short* __restrict__ dst,
                                 int K, int N, int row0, int ld) {
  int total = K * N;
  for (int i = blockIdx.x * blockDim.x + threadIdx.x; i < total;
       i += gridDim.x * blockDim.x) {
    int n = i / K, k = i - n * K;
    dst[i] = f2bf(src[(row0 + k) * ld + n]);
  }
}

// WoT: 16 x 512, rows 0..1 from Wo (512x2), rows 2..15 zero-padded.
__global__ void cast_wo_k(const float* __restrict__ Wo,
                          unsigned short* __restrict__ dst) {
  int i = blockIdx.x * blockDim.x + threadIdx.x;
  if (i < 16 * HD) {
    int n = i / HD, k = i - n * HD;
    float v = (n < 2) ? Wo[k * 2 + n] : 0.0f;
    dst[i] = f2bf(v);
  }
}

// ----------------------------- small GEMM ----------------------------------
// Y(M x N) = act(X(M x K, bf16) @ WT^T + bias), one wave per 16 x 128 tile.

template <int RELU, int OUTBF>
__global__ void __launch_bounds__(32) gemm_k(const unsigned short* __restrict__ X,
                                             const unsigned short* __restrict__ WT,
                                             const float* __restrict__ bias,
                                             void* __restrict__ Yv,
                                             int K, int N) {
  const int lane = threadIdx.x & 31;
  const int r  = lane & 15;
  const int hh = lane >> 4;
  const int mb  = blockIdx.x * 16;
  const int nb0 = blockIdx.y * 128;

  const v8f vzero = {0.f, 0.f, 0.f, 0.f, 0.f, 0.f, 0.f, 0.f};
  v8f acc[8];
#pragma unroll
  for (int t = 0; t < 8; ++t) acc[t] = vzero;

  const unsigned short* Abase = X + (size_t)(mb + r) * K + hh * 8;
  for (int k = 0; k < K; k += 32) {
    v16bf a = load_frag(Abase + k);
#pragma unroll
    for (int t = 0; t < 8; ++t) {
      v16bf w = load_frag(WT + (size_t)(nb0 + t * 16 + r) * K + hh * 8 + k);
      acc[t] = wmma_bf16(a, w, acc[t]);
    }
  }

#pragma unroll
  for (int t = 0; t < 8; ++t) {
    int col = nb0 + t * 16 + r;
    float bv = bias[col];
#pragma unroll
    for (int g = 0; g < 8; ++g) {
      int row = mb + hh * 8 + g;
      float v = acc[t][g] + bv;
      if (RELU) v = fmaxf(v, 0.0f);
      if (OUTBF) ((unsigned short*)Yv)[(size_t)row * N + col] = f2bf(v);
      else       ((float*)Yv)[(size_t)row * N + col] = v;
    }
  }
}

// ----------------------------- fused edge MLP -------------------------------
// WG-cooperative 512->512 layer.  128 threads / 4 waves; each wave owns 16
// rows.  W^T is staged in 128n x 64k tiles, double-buffered in LDS (wbuf),
// one s_barrier per stage; every staged byte is consumed by all 4 waves.
//   Awave : this wave's 16 rows, row-major stride HD (global F or LDS act)
//   outw  : this wave's 16-row slice of the LDS output activation buffer
//   WT    : global 512 x 512 bf16 (row n = output column, K contiguous)
//   wbuf  : LDS staging, 2 x (128 x 64) bf16 = 32 KB

static __device__ __forceinline__ void layer_coop(
    const unsigned short* Awave, unsigned short* outw,
    const unsigned short* __restrict__ WT, unsigned short* wbuf,
    const float* __restrict__ bias, int tid, int r, int hh) {
  const v8f vzero = {0.f, 0.f, 0.f, 0.f, 0.f, 0.f, 0.f, 0.f};
  const unsigned short* A0 = Awave + r * HD + hh * 8;

  // preload stage 0 (nb=0, kc=0): thread tid loads W^T row tid, k 0..63
  {
    const unsigned short* src = WT + (size_t)tid * HD;
    v4u pre[8];
#pragma unroll
    for (int q = 0; q < 8; ++q) pre[q] = *(const v4u*)(src + q * 8);
    unsigned short* dst = wbuf + tid * 64;
#pragma unroll
    for (int q = 0; q < 8; ++q) *(v4u*)(dst + q * 8) = pre[q];
  }
  __syncthreads();

  for (int nb = 0; nb < 4; ++nb) {
    v8f acc[8];
#pragma unroll
    for (int t = 0; t < 8; ++t) acc[t] = vzero;

    for (int kc = 0; kc < 8; ++kc) {
      const int s  = nb * 8 + kc;
      const int s1 = s + 1;

      // issue global loads for the next stage first (latency overlap)
      v4u nxt[8];
      if (s1 < 32) {
        const unsigned short* src =
            WT + (size_t)((s1 >> 3) * 128 + tid) * HD + (s1 & 7) * 64;
#pragma unroll
        for (int q = 0; q < 8; ++q) nxt[q] = *(const v4u*)(src + q * 8);
      }

      // WMMA burst from the staged tile (16 wmma per wave per stage)
      const unsigned short* wb = wbuf + (s & 1) * (128 * 64);
#pragma unroll
      for (int kk = 0; kk < 2; ++kk) {
        v16bf a = load_frag(A0 + kc * 64 + kk * 32);
#pragma unroll
        for (int t = 0; t < 8; ++t) {
          v16bf w = load_frag(wb + (t * 16 + r) * 64 + kk * 32 + hh * 8);
          acc[t] = wmma_bf16(a, w, acc[t]);
        }
      }

      // commit next stage to the other LDS buffer, then barrier
      if (s1 < 32) {
        unsigned short* dst = wbuf + (s1 & 1) * (128 * 64) + tid * 64;
#pragma unroll
        for (int q = 0; q < 8; ++q) *(v4u*)(dst + q * 8) = nxt[q];
      }
      __syncthreads();
    }

    // epilogue: bias + relu + bf16 into this wave's output rows
#pragma unroll
    for (int t = 0; t < 8; ++t) {
      int col = nb * 128 + t * 16 + r;
      float bv = bias[col];
#pragma unroll
      for (int g = 0; g < 8; ++g)
        outw[(hh * 8 + g) * HD + col] = f2bf(fmaxf(acc[t][g] + bv, 0.0f));
    }
  }
}

// grid = 2048 (bi*2 + jhalf), block = 128 (4 waves, 16 j-rows each).
// Dynamic LDS: act0(64KB) + act1(64KB) + wbuf(32KB) = 160KB -> 2 WG / WGP.
__global__ void __launch_bounds__(128) edge_k(
    const unsigned short* __restrict__ F,     // 1024 x 512 bf16 node features
    const float* __restrict__ C,              // 1024 x 512 f32: F@Wca_bot + bca
    const unsigned short* __restrict__ W1T,   // Wca_top^T 512x512
    const unsigned short* __restrict__ W2T,   // Wcb^T
    const unsigned short* __restrict__ W3T,   // Wcc^T
    const unsigned short* __restrict__ WoT,   // Wo^T padded 16x512
    const float* __restrict__ b2,
    const float* __restrict__ b3,
    const float* __restrict__ bo,
    float* __restrict__ out) {
  extern __shared__ unsigned short smem[];
  unsigned short* act0 = smem;                 // 64 x 512 bf16
  unsigned short* act1 = act0 + 64 * HD;       // 64 x 512 bf16
  unsigned short* wbuf = act1 + 64 * HD;       // 2 x 128 x 64 bf16

  const int tid  = threadIdx.x;
  const int wave = tid >> 5;
  const int lane = tid & 31;
  const int r  = lane & 15;
  const int hh = lane >> 4;

  const int wg = blockIdx.x;
  const int bi = wg >> 1;                       // b*128 + i
  const int b  = bi >> 7;
  const int j0 = ((wg & 1) << 6) + (wave << 4); // this wave's j base

  unsigned short* o0 = act0 + wave * 16 * HD;
  unsigned short* o1 = act1 + wave * 16 * HD;

  // Layer 1: A rows f[b, j0..j0+15] from global F; bias row = C[bi].
  layer_coop(F + (size_t)(b * NBRICK + j0) * HD, o0, W1T, wbuf,
             C + (size_t)bi * HD, tid, r, hh);
  // Layers 2 & 3: LDS -> LDS.
  layer_coop(o0, o1, W2T, wbuf, b2, tid, r, hh);
  layer_coop(o1, o0, W3T, wbuf, b3, tid, r, hh);

  // Layer 4: 512 -> 2 (N padded to 16); Wo^T is tiny, read straight from L2.
  {
    const v8f vzero = {0.f, 0.f, 0.f, 0.f, 0.f, 0.f, 0.f, 0.f};
    const unsigned short* A0 = o0 + r * HD + hh * 8;
    v8f acc = vzero;
    for (int k = 0; k < HD; k += 32) {
      v16bf w = load_frag(WoT + (size_t)r * HD + hh * 8 + k);
      acc = wmma_bf16(load_frag(A0 + k), w, acc);
    }
    if (r < 2) {
      float bv = bo[r];
#pragma unroll
      for (int g = 0; g < 8; ++g) {
        int j = j0 + hh * 8 + g;
        out[((size_t)bi * NBRICK + j) * 2 + r] = acc[g] + bv;
      }
    }
  }
}

// ----------------------------- launch --------------------------------------

extern "C" void kernel_launch(void* const* d_in, const int* in_sizes, int n_in,
                              void* d_out, int out_size, void* d_ws, size_t ws_size,
                              hipStream_t stream) {
  (void)in_sizes; (void)n_in; (void)out_size; (void)ws_size;

  const float* brick = (const float*)d_in[0];
  const float* Wa  = (const float*)d_in[1];
  const float* ba  = (const float*)d_in[2];
  const float* Wb  = (const float*)d_in[3];
  const float* bb  = (const float*)d_in[4];
  const float* Wca = (const float*)d_in[5];
  const float* bca = (const float*)d_in[6];
  const float* Wcb = (const float*)d_in[7];
  const float* bcb = (const float*)d_in[8];
  const float* Wcc = (const float*)d_in[9];
  const float* bcc = (const float*)d_in[10];
  const float* Wo  = (const float*)d_in[11];
  const float* bo  = (const float*)d_in[12];

  char* ws = (char*)d_ws;
  unsigned short* Xbf     = (unsigned short*)(ws + 0);        //  512 KB
  unsigned short* WaT     = (unsigned short*)(ws + 524288);   //  256 KB
  unsigned short* WbT     = (unsigned short*)(ws + 786432);   //  512 KB
  unsigned short* WcaTopT = (unsigned short*)(ws + 1310720);  //  512 KB
  unsigned short* WcaBotT = (unsigned short*)(ws + 1835008);  //  512 KB
  unsigned short* WcbT    = (unsigned short*)(ws + 2359296);  //  512 KB
  unsigned short* WccT    = (unsigned short*)(ws + 2883584);  //  512 KB
  unsigned short* WoT     = (unsigned short*)(ws + 3407872);  //   16 KB
  unsigned short* H1bf    = (unsigned short*)(ws + 3424256);  // 1024 KB
  unsigned short* Fbf     = (unsigned short*)(ws + 4472832);  // 1024 KB
  float*          Cmat    = (float*)         (ws + 5521408);  // 2048 KB

  // --- prep: casts + weight transposes (bf16) ---
  cast_k<<<256, 256, 0, stream>>>(brick, Xbf, MTOT * DIN);
  cast_transpose_k<<<128, 256, 0, stream>>>(Wa,  WaT,     DIN, HD, 0,   HD);
  cast_transpose_k<<<256, 256, 0, stream>>>(Wb,  WbT,     HD,  HD, 0,   HD);
  cast_transpose_k<<<256, 256, 0, stream>>>(Wca, WcaTopT, HD,  HD, 0,   HD);
  cast_transpose_k<<<256, 256, 0, stream>>>(Wca, WcaBotT, HD,  HD, 512, HD);
  cast_transpose_k<<<256, 256, 0, stream>>>(Wcb, WcbT,    HD,  HD, 0,   HD);
  cast_transpose_k<<<256, 256, 0, stream>>>(Wcc, WccT,    HD,  HD, 0,   HD);
  cast_wo_k<<<32, 256, 0, stream>>>(Wo, WoT);

  // --- node MLP + C = F @ Wca_bot + bca ---
  dim3 gg(MTOT / 16, HD / 128);
  gemm_k<1, 1><<<gg, 32, 0, stream>>>(Xbf,  WaT,     ba,  (void*)H1bf, DIN, HD);
  gemm_k<1, 1><<<gg, 32, 0, stream>>>(H1bf, WbT,     bb,  (void*)Fbf,  HD,  HD);
  gemm_k<0, 0><<<gg, 32, 0, stream>>>(Fbf,  WcaBotT, bca, (void*)Cmat, HD,  HD);

  // --- fused edge MLP: 2048 WGs x 128 threads, 160KB dynamic LDS ---
  edge_k<<<MTOT * 2, 128, 163840, stream>>>(Fbf, Cmat, WcaTopT, WcbT, WccT,
                                            WoT, bcb, bcc, bo, (float*)d_out);
}
